// GLN_10917806866600
// MI455X (gfx1250) — matrix operand order repaired and checked by reading, not compiled, not run on hardware
//
#include <hip/hip_runtime.h>
#include <math.h>

typedef __attribute__((ext_vector_type(16))) _Float16 v16h;
typedef __attribute__((ext_vector_type(8)))  float    v8f;

#define PCLIP 0.001f
#define L_LO  (-6.9067547786485535f)
#define L_HI  ( 6.9067547786485535f)

// Problem dims
#define BATCH   512
#define NCTX    1024
#define RTOT    2298      // 255*6 + 127*6 + 6 concatenated cm rows
#define RT_TILES 144      // rows padded to 2304 = 144 * 16
#define CT_TILES 32       // 512/16
#define KBLOCKS  32       // 1024/32

#define WMMA_F16(acc, A, B) \
    acc = __builtin_amdgcn_wmma_f32_16x16x32_f16(false, (A), false, (B), (short)0, (acc), false, false)

// ---------------------------------------------------------------------------
// Pack concatenated cm rows (f32 -> f16) into the WMMA A-matrix 16x32 per-lane
// layout (ISA 7.12.2): one thread per (row-tile, k-block, lane) writes 16
// halves as a single 32B chunk at  pA[(rt*32+kb)*512 + lane*16].
// ---------------------------------------------------------------------------
__global__ void prep_a(const float* __restrict__ cm0, const float* __restrict__ cm1,
                       const float* __restrict__ cm2, _Float16* __restrict__ pA) {
    int tid  = blockIdx.x * blockDim.x + threadIdx.x;
    int lane = tid & 31;
    int kb   = (tid >> 5) & 31;
    int rt   = tid >> 10;
    if (rt >= RT_TILES) return;
    int r = rt * 16 + (lane & 15);
    const float* src = nullptr;
    if      (r < 1530) src = cm0 + (size_t)r * 1024;
    else if (r < 2292) src = cm1 + (size_t)(r - 1530) * 1024;
    else if (r < RTOT) src = cm2 + (size_t)(r - 2292) * 1024;
    union { v16h v; _Float16 h[16]; } out;
    int lo8 = (lane < 16) ? 0 : 8;
    #pragma unroll
    for (int hh = 0; hh < 16; ++hh) {
        int vv = hh >> 1;
        int k  = ((vv < 4) ? 0 : 16) + lo8 + ((vv & 3) << 1) + (hh & 1);
        out.h[hh] = (_Float16)(src ? src[kb * 32 + k] : 0.0f);
    }
    *(v16h*)(pA + (size_t)(rt * 32 + kb) * 512 + lane * 16) = out.v;
}

// ---------------------------------------------------------------------------
// Pack x (f32 -> f16) into the WMMA B-matrix 32x16 per-lane layout:
// lane holds column N=lane&15; halves hh -> K = kb*32 + (hh<8?0:16) + (lane<16?0:8) + (hh&7).
// ---------------------------------------------------------------------------
__global__ void prep_b(const float* __restrict__ x, _Float16* __restrict__ pB) {
    int tid  = blockIdx.x * blockDim.x + threadIdx.x;
    int lane = tid & 31;
    int kb   = (tid >> 5) & 31;
    int ct   = tid >> 10;
    if (ct >= CT_TILES) return;
    int bcol  = ct * 16 + (lane & 15);
    int kbase = kb * 32 + ((lane < 16) ? 0 : 8);
    union { v16h v; _Float16 h[16]; } out;
    #pragma unroll
    for (int hh = 0; hh < 16; ++hh)
        out.h[hh] = (_Float16)x[(size_t)bcol * 1024 + kbase + ((hh < 8) ? 0 : 16) + (hh & 7)];
    *(v16h*)(pB + (size_t)(ct * 32 + kb) * 512 + lane * 16) = out.v;
}

// ---------------------------------------------------------------------------
// dist = cm_cat (2304x1024) @ x^T (1024x512).
// Each wave computes a 16x64 strip (1 row-tile x 4 col-tiles) using a manual
// 2-stage modulo schedule: two fixed register sets (even/odd k-chunk); each
// set is refilled right after its 4 WMMAs consume it, giving one chunk of
// load lookahead with no register rotation, no copies, and (~120 VGPRs) no
// spills. #pragma unroll 1 pins the schedule.
// ---------------------------------------------------------------------------
__global__ void gemm_dist(const _Float16* __restrict__ pA,
                          const _Float16* __restrict__ pB,
                          float* __restrict__ dist) {
    int w    = threadIdx.x >> 5;
    int lane = threadIdx.x & 31;
    int gw   = blockIdx.x * 8 + w;          // 1152 waves total
    int rt   = gw >> 3;                     // row tile   (0..143)
    int ctg  = gw & 7;                      // col group  (0..7), 4 tiles each
    const v16h* aP = (const v16h*)(pA + (size_t)rt * 32 * 512 + lane * 16);
    const v16h* bP = (const v16h*)(pB + (size_t)ctg * 4 * 32 * 512 + lane * 16);

    v8f acc0 = {}, acc1 = {}, acc2 = {}, acc3 = {};

    // prologue: set A <- chunk 0, set B <- chunk 1
    v16h a0 = aP[0];
    v16h p0 = bP[0], p1 = bP[1024], p2 = bP[2048], p3 = bP[3072];
    v16h a1 = aP[32];
    v16h q0 = bP[32], q1 = bP[32 + 1024], q2 = bP[32 + 2048], q3 = bP[32 + 3072];

    #pragma unroll 1
    for (int kb = 0; kb < KBLOCKS - 2; kb += 2) {
        // consume even chunk (set A), then refill it with chunk kb+2
        WMMA_F16(acc0, a0, p0);
        WMMA_F16(acc1, a0, p1);
        WMMA_F16(acc2, a0, p2);
        WMMA_F16(acc3, a0, p3);
        a0 = aP[(kb + 2) * 32];
        p0 = bP[(kb + 2) * 32];
        p1 = bP[(kb + 2) * 32 + 1024];
        p2 = bP[(kb + 2) * 32 + 2048];
        p3 = bP[(kb + 2) * 32 + 3072];
        // consume odd chunk (set B), then refill it with chunk kb+3
        WMMA_F16(acc0, a1, q0);
        WMMA_F16(acc1, a1, q1);
        WMMA_F16(acc2, a1, q2);
        WMMA_F16(acc3, a1, q3);
        a1 = aP[(kb + 3) * 32];
        q0 = bP[(kb + 3) * 32];
        q1 = bP[(kb + 3) * 32 + 1024];
        q2 = bP[(kb + 3) * 32 + 2048];
        q3 = bP[(kb + 3) * 32 + 3072];
    }
    // epilogue: chunks 30 (set A) and 31 (set B)
    WMMA_F16(acc0, a0, p0);
    WMMA_F16(acc1, a0, p1);
    WMMA_F16(acc2, a0, p2);
    WMMA_F16(acc3, a0, p3);
    WMMA_F16(acc0, a1, q0);
    WMMA_F16(acc1, a1, q1);
    WMMA_F16(acc2, a1, q2);
    WMMA_F16(acc3, a1, q3);

    int n  = lane & 15;
    int mb = (lane < 16) ? 0 : 8;
    union { v8f v; float f[8]; } c;
    c.v = acc0;
    #pragma unroll
    for (int j = 0; j < 8; ++j)
        dist[(size_t)(rt * 16 + mb + j) * BATCH + (ctg * 4 + 0) * 16 + n] = c.f[j];
    c.v = acc1;
    #pragma unroll
    for (int j = 0; j < 8; ++j)
        dist[(size_t)(rt * 16 + mb + j) * BATCH + (ctg * 4 + 1) * 16 + n] = c.f[j];
    c.v = acc2;
    #pragma unroll
    for (int j = 0; j < 8; ++j)
        dist[(size_t)(rt * 16 + mb + j) * BATCH + (ctg * 4 + 2) * 16 + n] = c.f[j];
    c.v = acc3;
    #pragma unroll
    for (int j = 0; j < 8; ++j)
        dist[(size_t)(rt * 16 + mb + j) * BATCH + (ctg * 4 + 3) * 16 + n] = c.f[j];
}

// 6 threshold bits -> gating index per (entry,b); entries: 255 L0, 127 L1, 1 L2.
__global__ void make_idx(const float* __restrict__ dist, const float* __restrict__ cb0,
                         const float* __restrict__ cb1, const float* __restrict__ cb2,
                         int* __restrict__ idxb) {
    int tid = blockIdx.x * blockDim.x + threadIdx.x;
    int b = tid & 511, e = tid >> 9;
    if (e >= 383) return;
    int rbase; const float* cb;
    if      (e < 255) { rbase = e * 6;                cb = cb0 + e * 6; }
    else if (e < 382) { int s = e - 255; rbase = 1530 + s * 6; cb = cb1 + s * 6; }
    else              { rbase = 2292;                 cb = cb2; }
    int idx = 0;
    #pragma unroll
    for (int m = 0; m < 6; ++m)
        idx |= (dist[(size_t)(rbase + m) * BATCH + b] > cb[m]) ? (1 << m) : 0;
    idxb[(size_t)e * BATCH + b] = idx;
}

__global__ void make_logits0(const float* __restrict__ x, const float* __restrict__ bb,
                             float* __restrict__ l0) {
    int tid = blockIdx.x * blockDim.x + threadIdx.x;
    int p = tid & 1023, b = tid >> 10;
    if (b >= BATCH) return;
    float v;
    if (p == 0) v = *bb;
    else {
        float cx = fminf(fmaxf(x[(size_t)b * 1024 + p], PCLIP), 1.0f - PCLIP);
        v = logf(cx / (1.0f - cx));
    }
    l0[(size_t)b * 1024 + p] = v;
}

// One block per batch row: logits row in LDS, each wave owns strided s values.
// Software-pipelined: next s's gating index is loaded early and its 4KB w-row
// prefetched (one cacheline per lane -> global_prefetch_b8) so the L2 gather
// latency hides behind the current dot product.
template <int PP, int SS, int EBASE>
__global__ void mix_layer(const float* __restrict__ lin, const float* __restrict__ wgt,
                          const int* __restrict__ idxb, const float* __restrict__ biasPtr,
                          float* __restrict__ lout, int outWidth) {
    __shared__ float lds[PP];
    int b = blockIdx.x;
    int lane = threadIdx.x & 31, w = threadIdx.x >> 5;
    for (int i = threadIdx.x; i < PP; i += blockDim.x) lds[i] = lin[(size_t)b * PP + i];
    __syncthreads();
    if (threadIdx.x == 0) lout[(size_t)b * outWidth] = *biasPtr;
    int idxCur = (w < SS) ? idxb[(size_t)(EBASE + w) * BATCH + b] : 0;
    for (int s = w; s < SS; s += 8) {
        const float* wrow = wgt + ((size_t)s * 64 + idxCur) * PP;
        int sn = s + 8;
        if (sn < SS) {
            idxCur = idxb[(size_t)(EBASE + sn) * BATCH + b];
            const char* nextRow = (const char*)(wgt + ((size_t)sn * 64 + idxCur) * PP);
            __builtin_prefetch(nextRow + (size_t)lane * 128, 0, 0);
        }
        float sum = 0.0f;
        #pragma unroll
        for (int t = 0; t < PP / 32; ++t)
            sum += wrow[lane + 32 * t] * lds[lane + 32 * t];
        #pragma unroll
        for (int off = 16; off; off >>= 1)
            sum += __shfl_xor(sum, off, 32);
        if (lane == 0)
            lout[(size_t)b * outWidth + 1 + s] = fminf(fmaxf(sum, L_LO), L_HI);
    }
}

__global__ void final_mix(const float* __restrict__ l2, const float* __restrict__ w2,
                          const int* __restrict__ idxb, float* __restrict__ out) {
    int lane = threadIdx.x & 31, w = threadIdx.x >> 5;
    int b = blockIdx.x * 8 + w;
    if (b >= BATCH) return;
    int idx = idxb[(size_t)382 * BATCH + b];
    const float* wrow = w2 + (size_t)idx * 128;
    float sum = 0.0f;
    #pragma unroll
    for (int t = 0; t < 4; ++t)
        sum += wrow[lane + 32 * t] * l2[(size_t)b * 128 + lane + 32 * t];
    #pragma unroll
    for (int off = 16; off; off >>= 1)
        sum += __shfl_xor(sum, off, 32);
    if (lane == 0) {
        float v = fminf(fmaxf(sum, L_LO), L_HI);
        out[b] = 1.0f / (1.0f + expf(-v));
    }
}

extern "C" void kernel_launch(void* const* d_in, const int* in_sizes, int n_in,
                              void* d_out, int out_size, void* d_ws, size_t ws_size,
                              hipStream_t stream) {
    const float* x    = (const float*)d_in[0];
    const float* bb   = (const float*)d_in[1];
    const float* cm0  = (const float*)d_in[2];
    const float* cb0  = (const float*)d_in[3];
    const float* w0   = (const float*)d_in[4];
    const float* b0   = (const float*)d_in[5];
    const float* cm1  = (const float*)d_in[6];
    const float* cb1  = (const float*)d_in[7];
    const float* w1   = (const float*)d_in[8];
    const float* b1   = (const float*)d_in[9];
    const float* cm2  = (const float*)d_in[10];
    const float* cb2  = (const float*)d_in[11];
    const float* w2   = (const float*)d_in[12];

    char* ws = (char*)d_ws;
    _Float16* pA  = (_Float16*)(ws + 0);         // 144*32*1024 B = 4,718,592
    _Float16* pB  = (_Float16*)(ws + 4718592);   // 32*32*1024 B  = 1,048,576
    float*    dist= (float*)(ws + 5767168);      // 2304*512*4    = 4,718,592
    int*      idxb= (int*)  (ws + 10485760);     // 383*512*4     =   784,384
    float*    l0  = (float*)(ws + 11270144);     // 512*1024*4    = 2,097,152
    float*    l1  = (float*)(ws + 13367296);     // 512*256*4     =   524,288
    float*    l2  = (float*)(ws + 13891584);     // 512*128*4     =   262,144
    float*    out = (float*)d_out;

    prep_a<<<576, 256, 0, stream>>>(cm0, cm1, cm2, pA);        // 144*32*32 threads
    prep_b<<<128, 256, 0, stream>>>(x, pB);                    // 32*32*32 threads
    make_logits0<<<2048, 256, 0, stream>>>(x, bb, l0);
    gemm_dist<<<144, 256, 0, stream>>>(pA, pB, dist);          // 1152 strip-waves
    make_idx<<<766, 256, 0, stream>>>(dist, cb0, cb1, cb2, idxb);
    mix_layer<1024, 255, 0><<<512, 256, 0, stream>>>(l0, w0, idxb, b0, l1, 256);
    mix_layer<256, 127, 255><<<512, 256, 0, stream>>>(l1, w1, idxb, b1, l2, 128);
    final_mix<<<64, 256, 0, stream>>>(l2, w2, idxb, out);
}
